// AttnBlock_80942953660753
// MI455X (gfx1250) — compile-verified, matrix operand order checked by reading, not run on hardware
//
#include <hip/hip_runtime.h>
#include <math.h>

// ---------------- problem constants (B=1) ----------------
constexpr int NT  = 16384;   // tokens
constexpr int CD  = 384;     // channels
constexpr int HH  = 8;       // heads
constexpr int HDD = 48;      // head dim
constexpr int KW  = 1024;    // window
constexpr int NWI = 16;      // windows

typedef _Float16 v16h __attribute__((ext_vector_type(16)));
typedef _Float16 v8h  __attribute__((ext_vector_type(8)));
typedef float    v8f  __attribute__((ext_vector_type(8)));

__device__ __forceinline__ v16h cat16(v8h lo, v8h hi) {
  return __builtin_shufflevector(lo, hi, 0,1,2,3,4,5,6,7,8,9,10,11,12,13,14,15);
}
__device__ __forceinline__ v8f wmma16(v16h a, v16h b, v8f c) {
  return __builtin_amdgcn_wmma_f32_16x16x32_f16(false, a, false, b, (short)0, c, false, false);
}

// Async global->LDS copy of 16 bytes per lane (CDNA5 GLOBAL_LOAD_ASYNC_TO_LDS_B128).
// ldsoff: per-lane LDS byte offset (generic shared addr low 32 bits), gaddr: per-lane global addr.
__device__ __forceinline__ void async_copy_b128(const void* lds_dst, const void* gsrc) {
  unsigned ldsoff = (unsigned)(size_t)lds_dst;
  asm volatile("global_load_async_to_lds_b128 %0, %1, off"
               :: "v"(ldsoff), "v"(gsrc) : "memory");
}
__device__ __forceinline__ void wait_async0() {
  asm volatile("s_wait_asynccnt 0" ::: "memory");
}

// ---------------- 1: per-axis min ----------------
__global__ __launch_bounds__(256) void k_min(const float* __restrict__ xyz, float* __restrict__ mins) {
  int ax = blockIdx.x;
  const float* p = xyz + (size_t)ax * NT;
  float m = 1e30f;
  for (int i = threadIdx.x; i < NT; i += 256) m = fminf(m, p[i]);
  __shared__ float sm[256];
  sm[threadIdx.x] = m; __syncthreads();
  for (int s = 128; s > 0; s >>= 1) {
    if (threadIdx.x < s) sm[threadIdx.x] = fminf(sm[threadIdx.x], sm[threadIdx.x + s]);
    __syncthreads();
  }
  if (threadIdx.x == 0) mins[ax] = sm[0];
}

// ---------------- 2: grid coords + 62-bit morton|idx keys ----------------
__device__ __forceinline__ unsigned interleave8(unsigned x, unsigned y, unsigned z) {
  unsigned key = 0;
  #pragma unroll
  for (int i = 0; i < 8; i++) {
    unsigned m = 1u << i;
    key |= ((x & m) << (2*i + 2)) | ((y & m) << (2*i + 1)) | ((z & m) << (2*i));
  }
  return key;
}
__global__ __launch_bounds__(256) void k_keys(const float* __restrict__ xyz, const float* __restrict__ mins,
                                              int* __restrict__ grid, unsigned long long* __restrict__ keys) {
  int i = blockIdx.x * 256 + threadIdx.x;
  if (i >= NT) return;
  unsigned g[3];
  #pragma unroll
  for (int a = 0; a < 3; a++) {
    float v = (xyz[(size_t)a * NT + i] - mins[a]) / 0.02f;
    int gi = (int)v;
    gi = gi < 0 ? 0 : (gi > 65535 ? 65535 : gi);
    grid[i * 3 + a] = gi;
    g[a] = (unsigned)gi;
  }
  unsigned lo = interleave8(g[0] & 255u, g[1] & 255u, g[2] & 255u);
  unsigned hi = interleave8((g[0] >> 8) & 255u, (g[1] >> 8) & 255u, (g[2] >> 8) & 255u);
  keys[i] = ((unsigned long long)hi << 38) | ((unsigned long long)lo << 14) | (unsigned long long)i;
}

// ---------------- 3: bitonic sort of 16384 u64 in LDS ----------------
__global__ __launch_bounds__(1024) void k_sort(const unsigned long long* __restrict__ keys, int* __restrict__ order) {
  extern __shared__ unsigned long long sk[];
  for (int i = threadIdx.x; i < NT; i += 1024) sk[i] = keys[i];
  __syncthreads();
  for (int k = 2; k <= NT; k <<= 1) {
    for (int j = k >> 1; j > 0; j >>= 1) {
      for (int t = threadIdx.x; t < NT / 2; t += 1024) {
        int i = 2 * t - (t & (j - 1));
        int ixj = i + j;
        bool up = ((i & k) == 0);
        unsigned long long a = sk[i], b = sk[ixj];
        if ((a > b) == up) { sk[i] = b; sk[ixj] = a; }
      }
      __syncthreads();
    }
  }
  for (int i = threadIdx.x; i < NT; i += 1024) order[i] = (int)(sk[i] & 0x3FFFull);
}

// ---------------- 4: gather by order + LN1 -> f32 shortcut, f16 xn ----------------
__global__ __launch_bounds__(128) void k_gather_ln1(const float* __restrict__ x, const int* __restrict__ order,
                                                    const float* __restrict__ g, const float* __restrict__ b,
                                                    float* __restrict__ shortcut, _Float16* __restrict__ xn) {
  int row = blockIdx.x;
  int o = order[row];
  __shared__ float red[128];
  float v[3];
  #pragma unroll
  for (int t = 0; t < 3; t++) { int c = threadIdx.x + 128 * t; v[t] = x[(size_t)c * NT + o]; }
  red[threadIdx.x] = v[0] + v[1] + v[2]; __syncthreads();
  for (int s = 64; s > 0; s >>= 1) { if (threadIdx.x < s) red[threadIdx.x] += red[threadIdx.x + s]; __syncthreads(); }
  float mu = red[0] / 384.f; __syncthreads();
  float d0 = v[0]-mu, d1 = v[1]-mu, d2 = v[2]-mu;
  red[threadIdx.x] = d0*d0 + d1*d1 + d2*d2; __syncthreads();
  for (int s = 64; s > 0; s >>= 1) { if (threadIdx.x < s) red[threadIdx.x] += red[threadIdx.x + s]; __syncthreads(); }
  float rstd = rsqrtf(red[0] / 384.f + 1e-5f);
  #pragma unroll
  for (int t = 0; t < 3; t++) {
    int c = threadIdx.x + 128 * t;
    shortcut[(size_t)row * CD + c] = v[t];
    xn[(size_t)row * CD + c] = (_Float16)((v[t] - mu) * rstd * g[c] + b[c]);
  }
}

// ---------------- 5: LN over rows (for LN2) ----------------
__global__ __launch_bounds__(128) void k_ln_rows(const float* __restrict__ xin, const float* __restrict__ g,
                                                 const float* __restrict__ b, _Float16* __restrict__ xout) {
  int row = blockIdx.x;
  __shared__ float red[128];
  float v[3];
  #pragma unroll
  for (int t = 0; t < 3; t++) { int c = threadIdx.x + 128 * t; v[t] = xin[(size_t)row * CD + c]; }
  red[threadIdx.x] = v[0] + v[1] + v[2]; __syncthreads();
  for (int s = 64; s > 0; s >>= 1) { if (threadIdx.x < s) red[threadIdx.x] += red[threadIdx.x + s]; __syncthreads(); }
  float mu = red[0] / 384.f; __syncthreads();
  float d0 = v[0]-mu, d1 = v[1]-mu, d2 = v[2]-mu;
  red[threadIdx.x] = d0*d0 + d1*d1 + d2*d2; __syncthreads();
  for (int s = 64; s > 0; s >>= 1) { if (threadIdx.x < s) red[threadIdx.x] += red[threadIdx.x + s]; __syncthreads(); }
  float rstd = rsqrtf(red[0] / 384.f + 1e-5f);
  #pragma unroll
  for (int t = 0; t < 3; t++) {
    int c = threadIdx.x + 128 * t;
    xout[(size_t)row * CD + c] = (_Float16)((v[t] - mu) * rstd * g[c] + b[c]);
  }
}

// ---------------- 6: f32 -> f16 weight convert ----------------
__global__ __launch_bounds__(256) void k_cvt(const float* __restrict__ in, _Float16* __restrict__ out, int n) {
  int i = blockIdx.x * 256 + threadIdx.x;
  if (i < n) out[i] = (_Float16)in[i];
}

// ---------------- 7: generic WMMA GEMM, 128x128x32 tiles, 8 waves ----------------
// EPI: 0 = bias -> f16 out        (QKV)
//      1 = bias + resid -> f32    (proj)
//      2 = bias + GELU -> f16     (FFN1)
//      3 = bias + resid -> f32 scatter out[col*NT + order[row]]  (FFN2 + unsort)
template <int EPI>
__global__ __launch_bounds__(256) void k_gemm(const _Float16* __restrict__ A, const _Float16* __restrict__ B,
                                              const float* __restrict__ bias, const float* __restrict__ resid,
                                              void* __restrict__ Out, const int* __restrict__ order,
                                              int Kd, int Nn) {
  __shared__ _Float16 ldsA[128 * 32];
  __shared__ _Float16 ldsBT[128 * 32];   // stored transposed: [n][k]
  int tid = threadIdx.x;
  int m0 = blockIdx.x * 128, n0 = blockIdx.y * 128;
  int w = tid >> 5, lane = tid & 31;
  int wm = w & 3, wn = w >> 2;           // 4 m-waves x 2 n-waves
  int hl = lane >> 4, l15 = lane & 15, kb = hl * 8;
  v8f acc[2][4];
  #pragma unroll
  for (int mi = 0; mi < 2; mi++)
    #pragma unroll
    for (int ni = 0; ni < 4; ni++) acc[mi][ni] = (v8f){};

  for (int k0 = 0; k0 < Kd; k0 += 32) {
    __syncthreads();
    #pragma unroll
    for (int it = 0; it < 2; it++) {           // A tile: 128x32 via async global->LDS b128
      int cid = tid + 256 * it;
      int r = cid >> 2, kc = (cid & 3) * 8;
      const _Float16* src = A + (size_t)(m0 + r) * Kd + k0 + kc;
      if (k0 + 32 < Kd) __builtin_prefetch(src + 32, 0, 3);
      async_copy_b128(&ldsA[r * 32 + kc], src);
    }
    #pragma unroll
    for (int it = 0; it < 2; it++) {           // B tile: 32x128, transposed into LDS
      int cid = tid + 256 * it;
      int kr = cid >> 4, nc = (cid & 15) * 8;
      v8h tmp = *(const v8h*)(B + (size_t)(k0 + kr) * Nn + n0 + nc);
      #pragma unroll
      for (int e = 0; e < 8; e++) ldsBT[(nc + e) * 32 + kr] = tmp[e];
    }
    wait_async0();
    __syncthreads();
    v16h af[2], bf[4];
    #pragma unroll
    for (int mi = 0; mi < 2; mi++) {
      int r = wm * 32 + mi * 16 + l15;
      af[mi] = cat16(*(const v8h*)&ldsA[r * 32 + kb], *(const v8h*)&ldsA[r * 32 + kb + 16]);
    }
    #pragma unroll
    for (int ni = 0; ni < 4; ni++) {
      int c = wn * 64 + ni * 16 + l15;
      bf[ni] = cat16(*(const v8h*)&ldsBT[c * 32 + kb], *(const v8h*)&ldsBT[c * 32 + kb + 16]);
    }
    #pragma unroll
    for (int mi = 0; mi < 2; mi++)
      #pragma unroll
      for (int ni = 0; ni < 4; ni++) acc[mi][ni] = wmma16(af[mi], bf[ni], acc[mi][ni]);
  }

  #pragma unroll
  for (int mi = 0; mi < 2; mi++)
    #pragma unroll
    for (int ni = 0; ni < 4; ni++)
      #pragma unroll
      for (int v = 0; v < 8; v++) {
        int row = m0 + wm * 32 + mi * 16 + v + 8 * hl;
        int col = n0 + wn * 64 + ni * 16 + l15;
        float val = acc[mi][ni][v] + bias[col];
        if constexpr (EPI == 0) {
          ((_Float16*)Out)[(size_t)row * Nn + col] = (_Float16)val;
        } else if constexpr (EPI == 1) {
          ((float*)Out)[(size_t)row * Nn + col] = val + resid[(size_t)row * Nn + col];
        } else if constexpr (EPI == 2) {
          val = 0.5f * val * (1.f + erff(val * 0.70710678f));
          ((_Float16*)Out)[(size_t)row * Nn + col] = (_Float16)val;
        } else {
          val += resid[(size_t)row * Nn + col];
          ((float*)Out)[(size_t)col * NT + order[row]] = val;
        }
      }
}

// ---------------- 8: RoPE + head split: qkv(N,1152) -> qh/kh/vh (H,N,48) ----------------
__global__ __launch_bounds__(256) void k_rope(const _Float16* __restrict__ qkv, const int* __restrict__ order,
                                              const int* __restrict__ grid,
                                              _Float16* __restrict__ qh, _Float16* __restrict__ kh,
                                              _Float16* __restrict__ vh) {
  int idx = blockIdx.x * 256 + threadIdx.x;
  if (idx >= NT * HH) return;
  int n = idx / HH, h = idx % HH;
  int o = order[n];
  const _Float16* base = qkv + (size_t)n * (3 * CD);
  size_t ob = ((size_t)h * NT + n) * HDD;
  #pragma unroll
  for (int d = 0; d < HDD; d++) vh[ob + d] = base[(2 * HH + h) * HDD + d];
  #pragma unroll
  for (int a = 0; a < 3; a++) {
    int p = grid[o * 3 + a];
    p = p < 0 ? 0 : (p > 4095 ? 4095 : p);
    float qv[16], kv[16];
    #pragma unroll
    for (int j = 0; j < 16; j++) {
      qv[j] = (float)base[(0 * HH + h) * HDD + a * 16 + j];
      kv[j] = (float)base[(1 * HH + h) * HDD + a * 16 + j];
    }
    #pragma unroll
    for (int j = 0; j < 16; j++) {
      float invf = __expf(-(float)(j & 7) * 0.57564627f);  // ln(100)/8
      float fr = (float)p * invf;
      float cs = __cosf(fr), sn = __sinf(fr);
      float rq = (j < 8) ? -qv[j + 8] : qv[j - 8];
      float rk = (j < 8) ? -kv[j + 8] : kv[j - 8];
      qh[ob + a * 16 + j] = (_Float16)(qv[j] * cs + rq * sn);
      kh[ob + a * 16 + j] = (_Float16)(kv[j] * cs + rk * sn);
    }
  }
}

// ---------------- 9: flash attention within 1024-token windows ----------------
// grid: H * NW * 8 blocks (q-tile of 128 rows); 8 waves of 16 q rows each; HD padded 48->64 in LDS.
__global__ __launch_bounds__(256) void k_attn(const _Float16* __restrict__ qh, const _Float16* __restrict__ kh,
                                              const _Float16* __restrict__ vh, _Float16* __restrict__ attn) {
  __shared__ _Float16 lq[128 * 64];
  __shared__ _Float16 lk[64 * 64];     // [key][d]
  __shared__ _Float16 lvt[64 * 64];    // [d][key]
  __shared__ _Float16 lp[8 * 16 * 64]; // per-wave P staging
  int bid = blockIdx.x;
  int h = bid >> 7;
  int wi = (bid >> 3) & 15;
  int qt = bid & 7;
  int n0 = wi * KW + qt * 128;
  int tid = threadIdx.x, w = tid >> 5, lane = tid & 31;
  int hl = lane >> 4, l15 = lane & 15, kb = hl * 8;

  // zero the 48..63 pad columns once (async copies below only touch d<48)
  for (int idx = tid; idx < 128 * 16; idx += 256) { int r = idx >> 4, c = 48 + (idx & 15); lq[r * 64 + c] = (_Float16)0.f; }
  if (tid < 64 * 16 / 4) {
    #pragma unroll
    for (int u = 0; u < 4; u++) {
      int idx = tid * 4 + u;
      int r = idx >> 4, c = 48 + (idx & 15);
      lk[r * 64 + c] = (_Float16)0.f;
    }
  }
  for (int idx = tid; idx < 16 * 64; idx += 256) { int d = 48 + (idx >> 6), c = idx & 63; lvt[d * 64 + c] = (_Float16)0.f; }

  // stage Q (128 rows x 48 halves) via async global->LDS, 16B chunks
  const _Float16* Qb = qh + ((size_t)h * NT + n0) * HDD;
  for (int cid = tid; cid < 128 * 6; cid += 256) {
    int r = cid / 6, c3 = cid % 6;
    async_copy_b128(&lq[r * 64 + c3 * 8], Qb + (size_t)r * HDD + c3 * 8);
  }
  wait_async0();
  __syncthreads();

  v16h aq[2];
  #pragma unroll
  for (int s = 0; s < 2; s++) {
    int r = w * 16 + l15;
    aq[s] = cat16(*(const v8h*)&lq[r * 64 + s * 32 + kb], *(const v8h*)&lq[r * 64 + s * 32 + kb + 16]);
  }

  float mrow[8], lrow[8];
  v8f o[4];
  #pragma unroll
  for (int ni = 0; ni < 4; ni++) o[ni] = (v8f){};
  #pragma unroll
  for (int v = 0; v < 8; v++) { mrow[v] = -1e30f; lrow[v] = 0.f; }
  const float scale = 0.14433756729740643f;  // 1/sqrt(48)

  for (int kc = 0; kc < 16; kc++) {
    __syncthreads();
    const _Float16* Kb = kh + ((size_t)h * NT + wi * KW + kc * 64) * HDD;
    const _Float16* Vb = vh + ((size_t)h * NT + wi * KW + kc * 64) * HDD;
    // K chunk (64 x 48) async into [key][d]
    for (int cid = tid; cid < 64 * 6; cid += 256) {
      int r = cid / 6, c3 = cid % 6;
      async_copy_b128(&lk[r * 64 + c3 * 8], Kb + (size_t)r * HDD + c3 * 8);
    }
    // V chunk transposed into [d][key]
    for (int idx = tid; idx < 64 * HDD; idx += 256) {
      int r = idx / HDD, c = idx % HDD;
      lvt[c * 64 + r] = Vb[(size_t)r * HDD + c];
    }
    wait_async0();
    __syncthreads();

    // S = Q K^T  (16 q rows x 64 keys per wave)
    v8f sacc[4];
    #pragma unroll
    for (int ni = 0; ni < 4; ni++) sacc[ni] = (v8f){};
    #pragma unroll
    for (int s = 0; s < 2; s++) {
      #pragma unroll
      for (int ni = 0; ni < 4; ni++) {
        int c = ni * 16 + l15;
        v16h bf = cat16(*(const v8h*)&lk[c * 64 + s * 32 + kb], *(const v8h*)&lk[c * 64 + s * 32 + kb + 16]);
        sacc[ni] = wmma16(aq[s], bf, sacc[ni]);
      }
    }

    // online softmax
    #pragma unroll
    for (int v = 0; v < 8; v++) {
      float mx = -1e30f;
      #pragma unroll
      for (int ni = 0; ni < 4; ni++) mx = fmaxf(mx, sacc[ni][v] * scale);
      #pragma unroll
      for (int d = 1; d < 16; d <<= 1) mx = fmaxf(mx, __shfl_xor(mx, d, 32));
      float mnew = fmaxf(mrow[v], mx);
      float corr = __expf(mrow[v] - mnew);
      lrow[v] *= corr;
      #pragma unroll
      for (int ni = 0; ni < 4; ni++) o[ni][v] *= corr;
      float psum = 0.f;
      int rl = v + 8 * hl;
      #pragma unroll
      for (int ni = 0; ni < 4; ni++) {
        float p = __expf(sacc[ni][v] * scale - mnew);
        psum += p;
        lp[(w * 16 + rl) * 64 + ni * 16 + l15] = (_Float16)p;
      }
      #pragma unroll
      for (int d = 1; d < 16; d <<= 1) psum += __shfl_xor(psum, d, 32);
      lrow[v] += psum;
      mrow[v] = mnew;
    }
    __syncthreads();

    // O += P V
    #pragma unroll
    for (int s = 0; s < 2; s++) {
      int r = w * 16 + l15;
      v16h ap = cat16(*(const v8h*)&lp[r * 64 + s * 32 + kb], *(const v8h*)&lp[r * 64 + s * 32 + kb + 16]);
      #pragma unroll
      for (int ni = 0; ni < 4; ni++) {
        int c = ni * 16 + l15;
        v16h bv = cat16(*(const v8h*)&lvt[c * 64 + s * 32 + kb], *(const v8h*)&lvt[c * 64 + s * 32 + kb + 16]);
        o[ni] = wmma16(ap, bv, o[ni]);
      }
    }
  }

  #pragma unroll
  for (int ni = 0; ni < 4; ni++)
    #pragma unroll
    for (int v = 0; v < 8; v++) {
      int d = ni * 16 + l15;
      if (d < HDD) {
        int row = n0 + w * 16 + v + 8 * hl;
        attn[(size_t)row * CD + h * HDD + d] = (_Float16)(o[ni][v] / lrow[v]);
      }
    }
}

// ---------------- launch ----------------
static size_t bump(size_t& p, size_t bytes) {
  size_t r = p;
  p += (bytes + 255) & ~(size_t)255;
  return r;
}

extern "C" void kernel_launch(void* const* d_in, const int* in_sizes, int n_in,
                              void* d_out, int out_size, void* d_ws, size_t ws_size,
                              hipStream_t stream) {
  const float* x      = (const float*)d_in[0];
  const float* xyz    = (const float*)d_in[1];
  const float* ln1_g  = (const float*)d_in[2];
  const float* ln1_b  = (const float*)d_in[3];
  const float* qkv_w  = (const float*)d_in[4];
  const float* qkv_b  = (const float*)d_in[5];
  const float* proj_w = (const float*)d_in[6];
  const float* proj_b = (const float*)d_in[7];
  const float* ln2_g  = (const float*)d_in[8];
  const float* ln2_b  = (const float*)d_in[9];
  const float* ffn_w1 = (const float*)d_in[10];
  const float* ffn_b1 = (const float*)d_in[11];
  const float* ffn_w2 = (const float*)d_in[12];
  const float* ffn_b2 = (const float*)d_in[13];

  char* ws = (char*)d_ws;
  size_t p = 0;
  float*              mins     = (float*)(ws + bump(p, 3 * 4));
  int*                grid     = (int*)(ws + bump(p, (size_t)NT * 3 * 4));
  unsigned long long* keys     = (unsigned long long*)(ws + bump(p, (size_t)NT * 8));
  int*                order    = (int*)(ws + bump(p, (size_t)NT * 4));
  float*              shortcut = (float*)(ws + bump(p, (size_t)NT * CD * 4));
  _Float16*           xn       = (_Float16*)(ws + bump(p, (size_t)NT * CD * 2));
  _Float16*           wqkvh    = (_Float16*)(ws + bump(p, (size_t)CD * 3 * CD * 2));
  _Float16*           wprojh   = (_Float16*)(ws + bump(p, (size_t)CD * CD * 2));
  _Float16*           w1h      = (_Float16*)(ws + bump(p, (size_t)CD * 4 * CD * 2));
  _Float16*           w2h      = (_Float16*)(ws + bump(p, (size_t)4 * CD * CD * 2));
  size_t off_qkvh = bump(p, (size_t)NT * 3 * CD * 2);
  _Float16*           qkvh     = (_Float16*)(ws + off_qkvh);
  _Float16*           qh       = (_Float16*)(ws + bump(p, (size_t)HH * NT * HDD * 2));
  _Float16*           kh       = (_Float16*)(ws + bump(p, (size_t)HH * NT * HDD * 2));
  _Float16*           vh       = (_Float16*)(ws + bump(p, (size_t)HH * NT * HDD * 2));
  _Float16*           attn     = (_Float16*)(ws + bump(p, (size_t)NT * CD * 2));
  float*              xt2      = (float*)(ws + bump(p, (size_t)NT * CD * 4));
  _Float16*           xn2      = (_Float16*)(ws + bump(p, (size_t)NT * CD * 2));
  _Float16*           ffnh     = (_Float16*)(ws + off_qkvh);  // aliases dead qkvh+qh (50 MB fits)

  // 1-3: serialization (z-order sort)
  k_min<<<3, 256, 0, stream>>>(xyz, mins);
  k_keys<<<(NT + 255) / 256, 256, 0, stream>>>(xyz, mins, grid, keys);
  k_sort<<<1, 1024, (size_t)NT * 8, stream>>>(keys, order);

  // 4: gather + LN1
  k_gather_ln1<<<NT, 128, 0, stream>>>(x, order, ln1_g, ln1_b, shortcut, xn);

  // weights -> f16
  k_cvt<<<(CD * 3 * CD + 255) / 256, 256, 0, stream>>>(qkv_w, wqkvh, CD * 3 * CD);
  k_cvt<<<(CD * CD + 255) / 256, 256, 0, stream>>>(proj_w, wprojh, CD * CD);
  k_cvt<<<(CD * 4 * CD + 255) / 256, 256, 0, stream>>>(ffn_w1, w1h, CD * 4 * CD);
  k_cvt<<<(4 * CD * CD + 255) / 256, 256, 0, stream>>>(ffn_w2, w2h, 4 * CD * CD);

  // 5: QKV GEMM (16384x384 @ 384x1152)
  k_gemm<0><<<dim3(NT / 128, (3 * CD) / 128), 256, 0, stream>>>(xn, wqkvh, qkv_b, nullptr, qkvh, nullptr, CD, 3 * CD);

  // 6: RoPE + head split
  k_rope<<<(NT * HH + 255) / 256, 256, 0, stream>>>(qkvh, order, grid, qh, kh, vh);

  // 7: windowed flash attention
  k_attn<<<HH * NWI * 8, 256, 0, stream>>>(qh, kh, vh, attn);

  // 8: proj GEMM + residual -> xt2 (f32)
  k_gemm<1><<<dim3(NT / 128, CD / 128), 256, 0, stream>>>(attn, wprojh, proj_b, shortcut, xt2, nullptr, CD, CD);

  // 9: LN2
  k_ln_rows<<<NT, 128, 0, stream>>>(xt2, ln2_g, ln2_b, xn2);

  // 10: FFN1 GEMM + GELU (16384x384 @ 384x1536)
  k_gemm<2><<<dim3(NT / 128, (4 * CD) / 128), 256, 0, stream>>>(xn2, w1h, ffn_b1, nullptr, ffnh, nullptr, CD, 4 * CD);

  // 11: FFN2 GEMM + residual + scatter-unsort straight into (C,N) output
  k_gemm<3><<<dim3(NT / 128, CD / 128), 256, 0, stream>>>(ffnh, w2h, ffn_b2, xt2, d_out, order, 4 * CD, CD);
}
